// SpeciesAwareLFMCModel_39178691674697
// MI455X (gfx1250) — matrix operand order repaired and checked by reading, not compiled, not run on hardware
//
#include <hip/hip_runtime.h>
#include <hip/hip_bf16.h>

typedef __attribute__((ext_vector_type(16))) _Float16 v16h;
typedef __attribute__((ext_vector_type(8)))  _Float16 v8h;
typedef __attribute__((ext_vector_type(2)))  _Float16 h2;
typedef __attribute__((ext_vector_type(8)))  float    v8f;

#define SPATIAL_LEVELS 24
#define TEMPORAL_LEVELS 19
#define SPATIAL_T  (1u << 22)
#define TEMPORAL_T (1u << 18)
#define XDIM  128   // 118 features padded to 128 (4 WMMA k-tiles)
#define H1DIM 256
#define H2DIM 128

// floor(16 * 1.45^l), l = 0..23  (computed in double, matches numpy float64 path)
__constant__ float c_spat_res[SPATIAL_LEVELS] = {
    16.f, 23.f, 33.f, 48.f, 70.f, 102.f, 148.f, 215.f, 312.f, 453.f, 657.f, 953.f,
    1382.f, 2004.f, 2905.f, 4213.f, 6109.f, 8858.f, 12845.f, 18625.f, 27007.f,
    39160.f, 56782.f, 82334.f};
// floor(8 * 1.45^l), l = 0..18
__constant__ float c_temp_res[TEMPORAL_LEVELS] = {
    8.f, 11.f, 16.f, 24.f, 35.f, 51.f, 74.f, 107.f, 156.f, 226.f, 328.f, 476.f,
    691.f, 1002.f, 1452.f, 2106.f, 3054.f, 4429.f, 6422.f};

// ---------------------------------------------------------------------------
// Kernel W: repack W1 (pad K 118->128) and W2 into WMMA B-fragment layout.
// B-fragment element e of lane L holds B[K = e + 16*(L/16)][N = L%16]
// (contiguous k order, matching the A loads in mlp_wmma). Flat layout: ((kt*NT + nt)*32 + lane)*16 + e
// so each lane's 16 halves are one contiguous 32-byte chunk.
// ---------------------------------------------------------------------------
__global__ void __launch_bounds__(256) pack_weights(
    const float* __restrict__ W1, const float* __restrict__ W2,
    _Float16* __restrict__ W1f, _Float16* __restrict__ W2f) {
  int tid = blockIdx.x * blockDim.x + threadIdx.x;
  if (tid < 32768) {  // W1: 4 ktiles x 16 ntiles x 32 lanes x 16 halves
    int e = tid & 15, lane = (tid >> 4) & 31, nt = (tid >> 9) & 15, kt = tid >> 13;
    int n = nt * 16 + (lane & 15);
    int K = kt * 32 + 16 * (lane >> 4) + e;
    W1f[tid] = (_Float16)(K < 118 ? W1[K * 256 + n] : 0.f);
  } else if (tid < 65536) {  // W2: 8 ktiles x 8 ntiles x 32 lanes x 16 halves
    int t = tid - 32768;
    int e = t & 15, lane = (t >> 4) & 31, nt = (t >> 9) & 7, kt = t >> 12;
    int n = nt * 16 + (lane & 15);
    int K = kt * 32 + 16 * (lane >> 4) + e;
    W2f[t] = (_Float16)W2[K * 128 + n];
  }
}

// ---------------------------------------------------------------------------
// Kernel A: Earth4D hash encode -> padded [N,128] f16 feature rows.
// Feature order: spatial 24 levels x2 | temporal 19x2 | species 32 | zero pad.
// ---------------------------------------------------------------------------
__global__ void __launch_bounds__(256) earth4d_encode(
    const float* __restrict__ coords, const int* __restrict__ species_idx,
    const float* __restrict__ spat_tab, const float* __restrict__ temp_tab,
    const float* __restrict__ species_emb, _Float16* __restrict__ Xf, int N) {
  int i = blockIdx.x * blockDim.x + threadIdx.x;
  if (i >= N) return;
  float4 c = ((const float4*)coords)[i];
  h2* row = (h2*)(Xf + (size_t)i * XDIM);

  // ---- spatial: trilinear interp of 8 hashed corners, per level ----
  const float2* tab = (const float2*)spat_tab;
  for (int l = 0; l < SPATIAL_LEVELS; ++l, tab += SPATIAL_T) {
    float res = c_spat_res[l];
    float px = c.x * res, py = c.y * res, pz = c.z * res;
    float fx = floorf(px), fy = floorf(py), fz = floorf(pz);
    unsigned ix = (unsigned)(int)fx, iy = (unsigned)(int)fy, iz = (unsigned)(int)fz;
    float tx = px - fx, ty = py - fy, tz = pz - fz;
    float f0 = 0.f, f1 = 0.f;
#pragma unroll
    for (int cn = 0; cn < 8; ++cn) {
      unsigned ox = (cn >> 2) & 1u, oy = (cn >> 1) & 1u, oz = cn & 1u;
      unsigned h = (ix + ox) ^ ((iy + oy) * 2654435761u) ^ ((iz + oz) * 805459861u);
      float2 f = tab[h & (SPATIAL_T - 1u)];
      float w = (ox ? tx : 1.f - tx) * (oy ? ty : 1.f - ty) * (oz ? tz : 1.f - tz);
      f0 += f.x * w;
      f1 += f.y * w;
    }
    row[l] = (h2){(_Float16)f0, (_Float16)f1};
  }

  // ---- temporal: linear interp of 2 corners, per level ----
  const float2* ttab = (const float2*)temp_tab;
  for (int l = 0; l < TEMPORAL_LEVELS; ++l, ttab += TEMPORAL_T) {
    float p = c.w * c_temp_res[l];
    float fp = floorf(p);
    float t = p - fp;
    unsigned i0 = (unsigned)(int)fp;
    float2 a = ttab[i0 & (TEMPORAL_T - 1u)];
    float2 b = ttab[(i0 + 1u) & (TEMPORAL_T - 1u)];
    row[SPATIAL_LEVELS + l] =
        (h2){(_Float16)((1.f - t) * a.x + t * b.x), (_Float16)((1.f - t) * a.y + t * b.y)};
  }

  // ---- species embedding gather (32 floats) ----
  const float* se = species_emb + (size_t)species_idx[i] * 32;
#pragma unroll
  for (int k = 0; k < 16; ++k)
    row[43 + k] = (h2){(_Float16)se[2 * k], (_Float16)se[2 * k + 1]};

  // ---- zero pad cols 118..127 ----
#pragma unroll
  for (int k = 0; k < 5; ++k) row[59 + k] = (h2){(_Float16)0.f, (_Float16)0.f};
}

// ---------------------------------------------------------------------------
// Kernel B: fused 3-layer MLP via v_wmma_f32_16x16x32_f16.
// 256 threads = 8 waves; each wave owns a 16-row tile.
// A-fragment from row-major f16: lane (hi=lane/16, lo=lane%16) loads two
// contiguous v8h at cols kt*32 + hi*16 and kt*32 + hi*16 + 8 of row `lo`
// (contiguous k order, matching the packed B fragments).
// ---------------------------------------------------------------------------
__global__ void __launch_bounds__(256, 1) mlp_wmma(
    const _Float16* __restrict__ Xf, const _Float16* __restrict__ W1f,
    const _Float16* __restrict__ W2f, const float* __restrict__ b1,
    const float* __restrict__ b2, const float* __restrict__ W3,
    const float* __restrict__ b3, float* __restrict__ out, int N) {
  __shared__ __align__(16) _Float16 lds[8][16][H1DIM];  // 64 KB: H1 then H2 staging
  const int lane = threadIdx.x & 31;
  const int w = threadIdx.x >> 5;
  const int hi = lane >> 4;
  const int lo = lane & 15;
  const int rowBase = blockIdx.x * 128 + w * 16;
  int row = rowBase + lo;
  if (row >= N) row = N - 1;  // tail clamp (stores guarded below)
  const _Float16* xrow = Xf + (size_t)row * XDIM;

  // ---------------- layer 1: [16x128] x [128x256] ----------------
  v8f acc[16];
#pragma unroll
  for (int nt = 0; nt < 16; ++nt) {
    float bv = b1[nt * 16 + lo];
    acc[nt] = (v8f){bv, bv, bv, bv, bv, bv, bv, bv};
  }
#pragma unroll
  for (int kt = 0; kt < 4; ++kt) {
    const _Float16* pa = xrow + kt * 32 + hi * 16;
    v8h alo = *(const v8h*)pa;
    v8h ahi = *(const v8h*)(pa + 8);
    v16h a = __builtin_shufflevector(alo, ahi, 0, 1, 2, 3, 4, 5, 6, 7, 8, 9, 10,
                                     11, 12, 13, 14, 15);
#pragma unroll
    for (int nt = 0; nt < 16; ++nt) {
      v16h b = *(const v16h*)(W1f + ((size_t)((kt * 16 + nt) * 32 + lane) * 16));
      acc[nt] = __builtin_amdgcn_wmma_f32_16x16x32_f16(false, a, false, b,
                                                       (short)0, acc[nt], false,
                                                       false);
    }
  }
  // ReLU -> stage H1 tile as f16 (D layout: lane holds row r+8*hi, col nt*16+lo)
#pragma unroll
  for (int nt = 0; nt < 16; ++nt)
#pragma unroll
    for (int r = 0; r < 8; ++r) {
      float v = acc[nt][r];
      lds[w][r + 8 * hi][nt * 16 + lo] = (_Float16)(v > 0.f ? v : 0.f);
    }
  __syncthreads();

  // ---------------- layer 2: [16x256] x [256x128] ----------------
  v8f acc2[8];
#pragma unroll
  for (int nt = 0; nt < 8; ++nt) {
    float bv = b2[nt * 16 + lo];
    acc2[nt] = (v8f){bv, bv, bv, bv, bv, bv, bv, bv};
  }
#pragma unroll
  for (int kt = 0; kt < 8; ++kt) {
    const _Float16* pa = &lds[w][lo][kt * 32 + hi * 16];
    v8h alo = *(const v8h*)pa;
    v8h ahi = *(const v8h*)(pa + 8);
    v16h a = __builtin_shufflevector(alo, ahi, 0, 1, 2, 3, 4, 5, 6, 7, 8, 9, 10,
                                     11, 12, 13, 14, 15);
#pragma unroll
    for (int nt = 0; nt < 8; ++nt) {
      v16h b = *(const v16h*)(W2f + ((size_t)((kt * 8 + nt) * 32 + lane) * 16));
      acc2[nt] = __builtin_amdgcn_wmma_f32_16x16x32_f16(false, a, false, b,
                                                        (short)0, acc2[nt],
                                                        false, false);
    }
  }
  __syncthreads();
  // ReLU -> stage H2 (reuse first 128 cols)
#pragma unroll
  for (int nt = 0; nt < 8; ++nt)
#pragma unroll
    for (int r = 0; r < 8; ++r) {
      float v = acc2[nt][r];
      lds[w][r + 8 * hi][nt * 16 + lo] = (_Float16)(v > 0.f ? v : 0.f);
    }
  __syncthreads();

  // ---------------- layer 3: [16x128] . [128] ----------------
  if (lane < 16) {
    float s = b3[0];
#pragma unroll 16
    for (int k = 0; k < H2DIM; ++k) s += (float)lds[w][lane][k] * W3[k];
    int orow = rowBase + lane;
    if (orow < N) out[orow] = s;
  }
}

// ---------------------------------------------------------------------------
extern "C" void kernel_launch(void* const* d_in, const int* in_sizes, int n_in,
                              void* d_out, int out_size, void* d_ws,
                              size_t ws_size, hipStream_t stream) {
  const float* coords      = (const float*)d_in[0];
  const int*   species_idx = (const int*)d_in[1];  // jax default x64-off -> i32
  const float* spat_tab    = (const float*)d_in[2];
  const float* temp_tab    = (const float*)d_in[3];
  const float* species_emb = (const float*)d_in[4];
  const float* W1 = (const float*)d_in[5];
  const float* b1 = (const float*)d_in[6];
  const float* W2 = (const float*)d_in[7];
  const float* b2 = (const float*)d_in[8];
  const float* W3 = (const float*)d_in[9];
  const float* b3 = (const float*)d_in[10];
  const int N = in_sizes[1];

  // workspace: [N,128] f16 features | 64KB W1 frags | 64KB W2 frags
  _Float16* Xf  = (_Float16*)d_ws;
  _Float16* W1f = Xf + (size_t)N * XDIM;
  _Float16* W2f = W1f + 32768;

  pack_weights<<<65536 / 256, 256, 0, stream>>>(W1, W2, W1f, W2f);
  earth4d_encode<<<(N + 255) / 256, 256, 0, stream>>>(
      coords, species_idx, spat_tab, temp_tab, species_emb, Xf, N);
  mlp_wmma<<<(N + 127) / 128, 256, 0, stream>>>(Xf, W1f, W2f, b1, b2, W3, b3,
                                                (float*)d_out, N);
}
